// Global_attention_Module_61718680044013
// MI455X (gfx1250) — matrix operand order repaired and checked by reading, not compiled, not run on hardware
//
#include <hip/hip_runtime.h>

#define D 128
#define BATCH 2
#define EPS_C 1e-6f
#define LN_EPS 1e-5f

typedef _Float16 v4h  __attribute__((ext_vector_type(4)));
typedef _Float16 v8h  __attribute__((ext_vector_type(8)));
typedef _Float16 v16h __attribute__((ext_vector_type(16)));
typedef float    v8f  __attribute__((ext_vector_type(8)));

union H16 { v16h v; v8h h[2]; };

static __device__ __forceinline__ v8f wmma_f16(v16h a, v16h b, v8f c) {
  // D(16x16 f32) = A(16x32 f16) * B(32x16 f16) + C
  return __builtin_amdgcn_wmma_f32_16x16x32_f16(false, a, false, b, (short)0, c,
                                                false, false);
}

// gfx1250 async global->LDS copy (16B per lane), tracked by ASYNCcnt.
static __device__ __forceinline__ void async_copy16(const void* g, const void* lds) {
  unsigned lo = (unsigned)(size_t)lds;                 // LDS aperture: low 32 bits = offset
  unsigned long long ga = (unsigned long long)(size_t)g;
  asm volatile("global_load_async_to_lds_b128 %0, %1, off"
               :: "v"(lo), "v"(ga) : "memory");
}
static __device__ __forceinline__ void wait_async0() {
#if __has_builtin(__builtin_amdgcn_s_wait_asynccnt)
  __builtin_amdgcn_s_wait_asynccnt(0);
#else
  asm volatile("s_wait_asynccnt 0x0" ::: "memory");
#endif
}

// ---------------------------------------------------------------- weight pre-swizzle
// packed[((p*8+nt)*4+ks)*32+lane][j] = W_p[k][n], n = nt*16+(lane&15),
// k = ks*32 + 8*(lane>>4) + (j<8 ? j : j+8)   (B-fragment layout of v_wmma_f32_16x16x32_f16)
__global__ void k_pack_w(const float* W0, const float* W1, const float* W2,
                         const float* W3, const float* W4, _Float16* __restrict__ pw) {
  int t = blockIdx.x * blockDim.x + threadIdx.x;
  if (t >= 5 * 8 * 4 * 32) return;
  int lane = t & 31, ks = (t >> 5) & 3, nt = (t >> 7) & 7, p = t >> 10;
  const float* W = (p == 0) ? W0 : (p == 1) ? W1 : (p == 2) ? W2 : (p == 3) ? W3 : W4;
  int n  = nt * 16 + (lane & 15);
  int kb = ks * 32 + 8 * (lane >> 4);
  _Float16* o = pw + (size_t)t * 16;
  for (int j = 0; j < 16; ++j) o[j] = (_Float16)W[(kb + (j < 8 ? j : j + 8)) * D + n];
}

__global__ void k_zero(float* p, long n) {
  long i = (long)blockIdx.x * blockDim.x + threadIdx.x;
  if (i < n) p[i] = 0.f;
}

// ---------------------------------------------------------------- one projection stage
// P==0: Q -> relu/sqrt(D) -> T        P==1: K -> kreg
// P==2: diag(K^T V), sum(K) atomics   P==3: raw -> T + LN stat atomics
// All per-tile addressing is strength-reduced: running column pointer (+16*Lpad),
// bias/atomic bases pre-offset by lane's lr so tile offsets are immediates.
template <int P>
static __device__ __forceinline__ void proj_stage(
    const _Float16* __restrict__ panel, const H16* a,
    const float* __restrict__ bias_lr,      // bias + lr
    int lane, _Float16* __restrict__ tcol,  // output column ptr at nt=0 (or nullptr)
    size_t colstride,                       // 16 * Lpad elements
    v8f* kreg, float* s0, float* s1,        // atomic bases + (b*D+lr) (or nullptr)
    bool full, const float* vmask, float inv_sc) {
#pragma unroll
  for (int nt = 0; nt < 8; ++nt) {
    const float bc = bias_lr[nt * 16];
    v8f acc;
#pragma unroll
    for (int r = 0; r < 8; ++r) acc[r] = bc;
    const _Float16* wp = panel + nt * 2048 + lane * 16;
#pragma unroll
    for (int ks = 0; ks < 4; ++ks) {
      H16 bf;
      bf.h[0] = *(const v8h*)(wp + ks * 512);
      bf.h[1] = *(const v8h*)(wp + ks * 512 + 8);
      acc = wmma_f16(a[ks].v, bf.v, acc);
    }
    if (P == 0) {
      v8h o;
#pragma unroll
      for (int r = 0; r < 8; ++r) o[r] = (_Float16)(fmaxf(acc[r], 0.f) * inv_sc);
      *(v8h*)tcol = o;
      tcol += colstride;
    } else if (P == 1) {
#pragma unroll
      for (int r = 0; r < 8; ++r) kreg[nt][r] = fmaxf(acc[r], 0.f) * inv_sc;
    } else if (P == 2) {
      float kv = 0.f, ksm = 0.f;
      if (full) {
#pragma unroll
        for (int r = 0; r < 8; ++r) { kv += kreg[nt][r] * acc[r]; ksm += kreg[nt][r]; }
      } else {
#pragma unroll
        for (int r = 0; r < 8; ++r) {
          kv += kreg[nt][r] * acc[r] * vmask[r];
          ksm += kreg[nt][r] * vmask[r];
        }
      }
      atomicAdd(&s0[nt * 16], kv);
      atomicAdd(&s1[nt * 16], ksm);
    } else {
      v8h o;
      float s = 0.f, q = 0.f;
      if (full) {
#pragma unroll
        for (int r = 0; r < 8; ++r) {
          float v = acc[r]; o[r] = (_Float16)v; s += v; q += v * v;
        }
      } else {
#pragma unroll
        for (int r = 0; r < 8; ++r) {
          float v = acc[r]; o[r] = (_Float16)v;
          s += v * vmask[r]; q += v * v * vmask[r];
        }
      }
      *(v8h*)tcol = o;
      tcol += colstride;
      atomicAdd(&s0[nt * 16], s);
      atomicAdd(&s1[nt * 16], q);
    }
  }
}

// ---------------------------------------------------------------- 5 fused projections
// h = x+pos computed on the fly; weight panels double-buffered in LDS via async
// global->LDS copies; 5 explicitly specialized stages (no runtime p-loop).
__global__ __launch_bounds__(256) void k_proj(
    const float* __restrict__ x, const float* __restrict__ pos,
    const _Float16* __restrict__ pw,
    const float* bq, const float* bk, const float* bv,
    const float* bq1, const float* bk1,
    _Float16* __restrict__ QT, _Float16* __restrict__ Q1T, _Float16* __restrict__ K1T,
    float* diagKV, float* Ksum, float* q1s, float* q1q, float* k1s, float* k1q,
    int L, int Lpad) {
  __shared__ alignas(16) _Float16 wlds[2][16384];  // ping-pong weight panels (2x32KB)
  const int tid = threadIdx.x, lane = tid & 31, wave = tid >> 5;
  const int lr = lane & 15, hi = lane >> 4;
  const int b = blockIdx.y;
  const int rowbase = blockIdx.x * 128 + wave * 16;
  const float inv_sc = rsqrtf((float)D);
  const bool full = (blockIdx.x * 128 + 128) <= L;   // block-uniform fast path

  // A fragments (16 rows x 128 K): read x/pos fp32, add, convert; reused by all 5 GEMMs
  H16 a[4];
  const int arow = rowbase + lr;
  if (arow < L) {
    const float* xp = x + ((size_t)b * L + arow) * D;
    const float* pp = pos + (size_t)arow * D;
#pragma unroll
    for (int ks = 0; ks < 4; ++ks) {
      int k0 = ks * 32 + hi * 8;
#pragma unroll
      for (int half = 0; half < 2; ++half) {
        int kk = k0 + half * 16;
        float4 x0 = *(const float4*)(xp + kk);
        float4 x1 = *(const float4*)(xp + kk + 4);
        float4 p0 = *(const float4*)(pp + kk);
        float4 p1 = *(const float4*)(pp + kk + 4);
        int jb = half * 8;
        a[ks].v[jb + 0] = (_Float16)(x0.x + p0.x);
        a[ks].v[jb + 1] = (_Float16)(x0.y + p0.y);
        a[ks].v[jb + 2] = (_Float16)(x0.z + p0.z);
        a[ks].v[jb + 3] = (_Float16)(x0.w + p0.w);
        a[ks].v[jb + 4] = (_Float16)(x1.x + p1.x);
        a[ks].v[jb + 5] = (_Float16)(x1.y + p1.y);
        a[ks].v[jb + 6] = (_Float16)(x1.z + p1.z);
        a[ks].v[jb + 7] = (_Float16)(x1.w + p1.w);
      }
    }
  } else {
#pragma unroll
    for (int ks = 0; ks < 4; ++ks)
#pragma unroll
      for (int j = 0; j < 16; ++j) a[ks].v[j] = (_Float16)0.f;
  }

  float vmask[8];
#pragma unroll
  for (int r = 0; r < 8; ++r)
    vmask[r] = (rowbase + r + 8 * hi) < L ? 1.0f : 0.0f;

  auto stage_panel = [&](int p, int buf) {
    const char* src = (const char*)(pw + (size_t)p * 16384);
    char* dst = (char*)wlds[buf];
#pragma unroll
    for (int i = 0; i < 8; ++i) {
      int off = (i * 256 + tid) * 16;
      async_copy16(src + off, dst + off);
    }
  };

  // hoisted per-stage bases (one 64-bit multiply each, not per tile)
  const size_t colstride = (size_t)16 * Lpad;
  const size_t tbase = (size_t)(b * D + lr) * Lpad + rowbase + 8 * hi;
  const int slr = b * D + lr;

  v8f kreg[8];  // K tile kept in registers between stages 1 and 2

  // explicit double-buffered pipeline: copy p+1 while computing p
  stage_panel(0, 0);
  wait_async0();
  __syncthreads();
  stage_panel(1, 1);
  proj_stage<0>(wlds[0], a, bq + lr, lane, QT + tbase, colstride,
                nullptr, nullptr, nullptr, full, vmask, inv_sc);
  wait_async0();
  __syncthreads();        // buf1 ready; all waves done reading buf0
  stage_panel(2, 0);
  proj_stage<1>(wlds[1], a, bk + lr, lane, nullptr, colstride,
                kreg, nullptr, nullptr, full, vmask, inv_sc);
  wait_async0();
  __syncthreads();
  stage_panel(3, 1);
  proj_stage<2>(wlds[0], a, bv + lr, lane, nullptr, colstride,
                kreg, diagKV + slr, Ksum + slr, full, vmask, inv_sc);
  wait_async0();
  __syncthreads();
  stage_panel(4, 0);
  proj_stage<3>(wlds[1], a, bq1 + lr, lane, Q1T + tbase, colstride,
                nullptr, q1s + slr, q1q + slr, full, vmask, inv_sc);
  wait_async0();
  __syncthreads();
  proj_stage<3>(wlds[0], a, bk1 + lr, lane, K1T + tbase, colstride,
                nullptr, k1s + slr, k1q + slr, full, vmask, inv_sc);
}

// ---------------------------------------------------------------- finalize stats
__global__ void k_stats(const float* q1s, const float* q1q, const float* k1s,
                        const float* k1q, const float* Ksum,
                        float* muq, float* rsq, float* muk, float* rsk,
                        float* kse, int L) {
  int i = blockIdx.x * blockDim.x + threadIdx.x;
  if (i >= BATCH * D) return;
  float invL = 1.0f / (float)L;
  float m = q1s[i] * invL;
  muq[i] = m;
  rsq[i] = rsqrtf(fmaxf(q1q[i] * invL - m * m, 0.f) + LN_EPS);
  m = k1s[i] * invL;
  muk[i] = m;
  rsk[i] = rsqrtf(fmaxf(k1q[i] * invL - m * m, 0.f) + LN_EPS);
  kse[i] = Ksum[i] + EPS_C;
}

// ---------------------------------------------------------------- S = Q1n^T K1n (split-K)
__global__ __launch_bounds__(256) void k_qk1(
    const _Float16* __restrict__ Q1T, const _Float16* __restrict__ K1T,
    const float* muq, const float* rsq, const float* muk, const float* rsk,
    const float* gq, const float* bgq, const float* gk, const float* bgk,
    float* __restrict__ S, int L, int Lpad, int chunks) {
  __shared__ alignas(16) _Float16 QTs[128 * 32];  // [c][l] LN+relu applied
  __shared__ alignas(16) _Float16 KTs[128 * 32];
  const int tid = threadIdx.x, lane = tid & 31, wave = tid >> 5;
  const int lr = lane & 15, hi = lane >> 4;
  const int b = blockIdx.x / chunks;
  const int chunk = blockIdx.x % chunks;
  const float inv_sL = rsqrtf((float)L);

  const int sc = tid >> 1, llb = (tid & 1) * 16;
  const float muqc = muq[b * D + sc], rsqc = rsq[b * D + sc];
  const float mukc = muk[b * D + sc], rskc = rsk[b * D + sc];
  const _Float16* qrow = Q1T + (size_t)(b * D + sc) * Lpad;
  const _Float16* krow = K1T + (size_t)(b * D + sc) * Lpad;

  v8f acc[8];
#pragma unroll
  for (int i = 0; i < 8; ++i)
#pragma unroll
    for (int r = 0; r < 8; ++r) acc[i][r] = 0.f;

  H16 qr, kr;  // raw chunk in flight
  {
    const int lg = chunk * 256 + llb;
    qr.h[0] = *(const v8h*)(qrow + lg);
    qr.h[1] = *(const v8h*)(qrow + lg + 8);
    kr.h[0] = *(const v8h*)(krow + lg);
    kr.h[1] = *(const v8h*)(krow + lg + 8);
  }

  for (int ks = 0; ks < 8; ++ks) {
    const int lg = chunk * 256 + ks * 32 + llb;
    __syncthreads();
    {
      H16 qo, ko;
      if (lg + 16 <= L) {  // fast unmasked path (all but the last chunk)
        float gqv[16], bqv[16], gkv[16], bkv[16];
#pragma unroll
        for (int j4 = 0; j4 < 4; ++j4) {
          float4 t;
          t = *(const float4*)(gq  + lg + j4 * 4);
          gqv[j4*4] = t.x; gqv[j4*4+1] = t.y; gqv[j4*4+2] = t.z; gqv[j4*4+3] = t.w;
          t = *(const float4*)(bgq + lg + j4 * 4);
          bqv[j4*4] = t.x; bqv[j4*4+1] = t.y; bqv[j4*4+2] = t.z; bqv[j4*4+3] = t.w;
          t = *(const float4*)(gk  + lg + j4 * 4);
          gkv[j4*4] = t.x; gkv[j4*4+1] = t.y; gkv[j4*4+2] = t.z; gkv[j4*4+3] = t.w;
          t = *(const float4*)(bgk + lg + j4 * 4);
          bkv[j4*4] = t.x; bkv[j4*4+1] = t.y; bkv[j4*4+2] = t.z; bkv[j4*4+3] = t.w;
        }
#pragma unroll
        for (int j = 0; j < 16; ++j) {
          qo.v[j] = (_Float16)(fmaxf(((float)qr.v[j] - muqc) * rsqc * gqv[j] + bqv[j], 0.f) * inv_sL);
          ko.v[j] = (_Float16)(fmaxf(((float)kr.v[j] - mukc) * rskc * gkv[j] + bkv[j], 0.f) * inv_sL);
        }
      } else {             // masked tail
#pragma unroll
        for (int j = 0; j < 16; ++j) {
          int l = lg + j;
          float qv = 0.f, kv = 0.f;
          if (l < L) {
            qv = fmaxf(((float)qr.v[j] - muqc) * rsqc * gq[l] + bgq[l], 0.f) * inv_sL;
            kv = fmaxf(((float)kr.v[j] - mukc) * rskc * gk[l] + bgk[l], 0.f) * inv_sL;
          }
          qo.v[j] = (_Float16)qv;
          ko.v[j] = (_Float16)kv;
        }
      }
      *(v8h*)(QTs + sc * 32 + llb)     = qo.h[0];
      *(v8h*)(QTs + sc * 32 + llb + 8) = qo.h[1];
      *(v8h*)(KTs + sc * 32 + llb)     = ko.h[0];
      *(v8h*)(KTs + sc * 32 + llb + 8) = ko.h[1];
    }
    __syncthreads();

    if (ks < 7) {  // issue next chunk's loads; they overlap the WMMAs below
      const int ln = lg + 32;
      qr.h[0] = *(const v8h*)(qrow + ln);
      qr.h[1] = *(const v8h*)(qrow + ln + 8);
      kr.h[0] = *(const v8h*)(krow + ln);
      kr.h[1] = *(const v8h*)(krow + ln + 8);
    }

    H16 af;  // A row c = wave*16+lr, K dim = l
    const _Float16* qp = QTs + (wave * 16 + lr) * 32 + hi * 8;
    af.h[0] = *(const v8h*)qp;
    af.h[1] = *(const v8h*)(qp + 16);
#pragma unroll
    for (int dt = 0; dt < 8; ++dt) {
      H16 bf;  // B col d = dt*16+lr
      const _Float16* kp = KTs + (dt * 16 + lr) * 32 + hi * 8;
      bf.h[0] = *(const v8h*)kp;
      bf.h[1] = *(const v8h*)(kp + 16);
      acc[dt] = wmma_f16(af.v, bf.v, acc[dt]);
    }
  }
#pragma unroll
  for (int dt = 0; dt < 8; ++dt)
#pragma unroll
    for (int r = 0; r < 8; ++r) {
      int c = wave * 16 + r + 8 * hi;
      int d = dt * 16 + lr;
      atomicAdd(&S[((size_t)b * D + c) * D + d], acc[dt][r]);
    }
}

// ---------------------------------------------------------------- softmax -> midT (f16, [d][c])
__global__ void k_softmax(const float* __restrict__ S, _Float16* __restrict__ midT) {
  __shared__ float red[128];
  const int row = blockIdx.x;  // over B*D rows (b,c)
  const int b = row >> 7, c = row & 127;
  const int t = threadIdx.x;   // d
  float v = S[(size_t)row * D + t];
  red[t] = v;
  __syncthreads();
  for (int s = 64; s > 0; s >>= 1) {
    if (t < s) red[t] = fmaxf(red[t], red[t + s]);
    __syncthreads();
  }
  float m = red[0];
  __syncthreads();
  float e = __expf(v - m);
  red[t] = e;
  __syncthreads();
  for (int s = 64; s > 0; s >>= 1) {
    if (t < s) red[t] += red[t + s];
    __syncthreads();
  }
  midT[(size_t)b * D * D + t * D + c] = (_Float16)(e / red[0]);  // transposed store
}

// ---------------------------------------------------------------- fused epilogue
__global__ __launch_bounds__(256) void k_out(
    const _Float16* __restrict__ QT, const _Float16* __restrict__ midTg,
    const float* __restrict__ x, const float* diagKV, const float* kse,
    const float* gamma_p, const float* g_out, const float* b_out,
    float* __restrict__ out, int L, int Lpad, int rowBlocks) {
  __shared__ alignas(16) char smem[49152];
  _Float16* midT = (_Float16*)smem;             // [d][c] 32KB
  _Float16* qls  = (_Float16*)(smem + 32768);   // [64 rows][128 c] 16KB; becomes r in-place
  float* olds = (float*)smem;                   // [64][128] f32, reuses midT region
  __shared__ float dkv[128], ksel[128];

  const int tid = threadIdx.x, lane = tid & 31, wave = tid >> 5;
  const int lr = lane & 15, hi = lane >> 4;
  const int b = blockIdx.x / rowBlocks;
  const int rowbase = (blockIdx.x % rowBlocks) * 64;
  const float gamma = gamma_p[0];

  if (tid < 128) { dkv[tid] = diagKV[b * D + tid]; ksel[tid] = kse[b * D + tid]; }

  // async contiguous 32KB copy of the transposed mid panel into LDS
  {
    const char* mp = (const char*)(midTg + (size_t)b * D * D);
#pragma unroll
    for (int i = 0; i < 8; ++i) {
      int off = (i * 256 + tid) * 16;
      async_copy16(mp + off, (const char*)midT + off);
    }
  }
  // stage Q rows [rowbase..rowbase+64) from transposed QT, transposing into LDS
  {
    const int c = tid >> 1;
    const int lb = (tid & 1) * 32;
    const _Float16* qp = QT + (size_t)(b * D + c) * Lpad + rowbase + lb;
#pragma unroll
    for (int i = 0; i < 4; ++i) {
      v8h q8 = *(const v8h*)(qp + i * 8);
#pragma unroll
      for (int j = 0; j < 8; ++j) qls[(lb + i * 8 + j) * D + c] = q8[j];
    }
  }
  wait_async0();
  __syncthreads();

  // Phase A: r = Q * diagKV * Z, per row, written in place over qls (f16)
  if (tid < 64) {
    int row = rowbase + tid;
    if (row < L) {
      float dot = 0.f;
      for (int c = 0; c < D; c += 8) {
        v8h q8 = *(const v8h*)(qls + tid * D + c);
#pragma unroll
        for (int j = 0; j < 8; ++j) dot += (float)q8[j] * ksel[c + j];
      }
      float z = 1.0f / dot;
      for (int c = 0; c < D; c += 8) {
        v8h q8 = *(const v8h*)(qls + tid * D + c);
        v8h r8;
#pragma unroll
        for (int j = 0; j < 8; ++j) r8[j] = (_Float16)((float)q8[j] * dkv[c + j] * z);
        *(v8h*)(qls + tid * D + c) = r8;
      }
    } else {
      for (int c = 0; c < D; c += 8) {
        v8h z8;
#pragma unroll
        for (int j = 0; j < 8; ++j) z8[j] = (_Float16)0.f;
        *(v8h*)(qls + tid * D + c) = z8;
      }
    }
  }
  __syncthreads();

  // Phase B: out = r @ mid via WMMA (4 row-tiles x 8 col-tiles, 4 tiles/wave)
  const int rt = wave & 3;
  v8f acc[4];
#pragma unroll
  for (int i = 0; i < 4; ++i)
#pragma unroll
    for (int r = 0; r < 8; ++r) acc[i][r] = 0.f;
#pragma unroll
  for (int kb = 0; kb < 128; kb += 32) {
    H16 af;
    const _Float16* rp = qls + (rt * 16 + lr) * D + kb + hi * 8;
    af.h[0] = *(const v8h*)rp;
    af.h[1] = *(const v8h*)(rp + 16);
#pragma unroll
    for (int i = 0; i < 4; ++i) {
      int dt = (wave >> 2) * 4 + i;
      H16 bf;
      const _Float16* mpp = midT + (dt * 16 + lr) * D + kb + hi * 8;
      bf.h[0] = *(const v8h*)mpp;
      bf.h[1] = *(const v8h*)(mpp + 16);
      acc[i] = wmma_f16(af.v, bf.v, acc[i]);
    }
  }
  __syncthreads();  // all LDS reads done; safe to overwrite midT region
#pragma unroll
  for (int i = 0; i < 4; ++i) {
    int dt = (wave >> 2) * 4 + i;
#pragma unroll
    for (int r = 0; r < 8; ++r)
      olds[(rt * 16 + r + 8 * hi) * D + dt * 16 + lr] = acc[i][r];
  }
  __syncthreads();

  // Phase C: residual + LayerNorm over D, vectorized f32x4 stores
  if (tid < 64) {
    int row = rowbase + tid;
    if (row < L) {
      const float* xp = x + ((size_t)b * L + row) * D;
      float* op = out + ((size_t)b * L + row) * D;
      float s = 0.f, s2 = 0.f;
      for (int c = 0; c < D; c += 4) {
        float4 xv = *(const float4*)(xp + c);
        float4 o;
        o.x = xv.x + gamma * olds[tid * D + c];
        o.y = xv.y + gamma * olds[tid * D + c + 1];
        o.z = xv.z + gamma * olds[tid * D + c + 2];
        o.w = xv.w + gamma * olds[tid * D + c + 3];
        olds[tid * D + c]     = o.x;
        olds[tid * D + c + 1] = o.y;
        olds[tid * D + c + 2] = o.z;
        olds[tid * D + c + 3] = o.w;
        s += o.x + o.y + o.z + o.w;
        s2 += o.x * o.x + o.y * o.y + o.z * o.z + o.w * o.w;
      }
      float mu = s * (1.0f / (float)D);
      float rstd = rsqrtf(s2 * (1.0f / (float)D) - mu * mu + LN_EPS);
      for (int c = 0; c < D; c += 4) {
        float4 o;
        o.x = (olds[tid * D + c]     - mu) * rstd * g_out[c]     + b_out[c];
        o.y = (olds[tid * D + c + 1] - mu) * rstd * g_out[c + 1] + b_out[c + 1];
        o.z = (olds[tid * D + c + 2] - mu) * rstd * g_out[c + 2] + b_out[c + 2];
        o.w = (olds[tid * D + c + 3] - mu) * rstd * g_out[c + 3] + b_out[c + 3];
        *(float4*)(op + c) = o;
      }
    }
  }
}

// ---------------------------------------------------------------- launcher
extern "C" void kernel_launch(void* const* d_in, const int* in_sizes, int n_in,
                              void* d_out, int out_size, void* d_ws, size_t ws_size,
                              hipStream_t stream) {
  const float* x    = (const float*)d_in[0];
  const float* pos  = (const float*)d_in[1];
  const float* Wq   = (const float*)d_in[2];
  const float* bq   = (const float*)d_in[3];
  const float* Wk   = (const float*)d_in[4];
  const float* bk   = (const float*)d_in[5];
  const float* Wv   = (const float*)d_in[6];
  const float* bv   = (const float*)d_in[7];
  const float* Wq1  = (const float*)d_in[8];
  const float* bq1  = (const float*)d_in[9];
  const float* Wk1  = (const float*)d_in[10];
  const float* bk1  = (const float*)d_in[11];
  const float* gq1  = (const float*)d_in[12];
  const float* bgq1 = (const float*)d_in[13];
  const float* gk1  = (const float*)d_in[14];
  const float* bgk1 = (const float*)d_in[15];
  const float* gam  = (const float*)d_in[16];
  const float* gout = (const float*)d_in[17];
  const float* bout = (const float*)d_in[18];

  const int L = in_sizes[0] / (BATCH * D);
  const int Lpad = ((L + 255) / 256) * 256;
  char* ws = (char*)d_ws;
  size_t off = 0;
  auto alloc = [&](size_t bytes) {
    size_t o = off;
    off += (bytes + 255) & ~(size_t)255;
    return o;
  };
  _Float16* QT   = (_Float16*)(ws + alloc((size_t)BATCH * D * Lpad * 2));
  _Float16* Q1T  = (_Float16*)(ws + alloc((size_t)BATCH * D * Lpad * 2));
  _Float16* K1T  = (_Float16*)(ws + alloc((size_t)BATCH * D * Lpad * 2));
  _Float16* pw   = (_Float16*)(ws + alloc((size_t)5 * 1024 * 16 * 2));
  float*    small= (float*)(ws + alloc((size_t)(11 * 256 + BATCH * D * D) * 4));
  _Float16* midb = (_Float16*)(ws + alloc((size_t)BATCH * D * D * 2));

  float* diagKV = small;
  float* Ksum   = small + 256;
  float* kse    = small + 512;
  float* q1s    = small + 768;
  float* q1q    = small + 1024;
  float* k1s    = small + 1280;
  float* k1q    = small + 1536;
  float* muq    = small + 1792;
  float* rsq    = small + 2048;
  float* muk    = small + 2304;
  float* rsk    = small + 2560;
  float* S      = small + 2816;

  k_pack_w<<<20, 256, 0, stream>>>(Wq, Wk, Wv, Wq1, Wk1, pw);
  const long nz = 2816 + (long)BATCH * D * D;
  k_zero<<<(unsigned)((nz + 255) / 256), 256, 0, stream>>>(small, nz);

  dim3 g2((L + 127) / 128, BATCH);
  k_proj<<<g2, 256, 0, stream>>>(x, pos, pw, bq, bk, bv, bq1, bk1,
                                 QT, Q1T, K1T,
                                 diagKV, Ksum, q1s, q1q, k1s, k1q, L, Lpad);
  k_stats<<<1, 256, 0, stream>>>(q1s, q1q, k1s, k1q, Ksum, muq, rsq, muk, rsk, kse, L);

  const int chunks = Lpad / 256;
  k_qk1<<<BATCH * chunks, 256, 0, stream>>>(Q1T, K1T, muq, rsq, muk, rsk,
                                            gq1, bgq1, gk1, bgk1, S, L, Lpad, chunks);
  k_softmax<<<BATCH * D, 128, 0, stream>>>(S, midb);

  const int rowBlocks = (L + 63) / 64;
  k_out<<<BATCH * rowBlocks, 256, 0, stream>>>(QT, midb, x, diagKV, kse,
                                               gam, gout, bout, (float*)d_out, L, Lpad, rowBlocks);
}
